// GNNEdgeClassifier_17978733101709
// MI455X (gfx1250) — compile-verified, matrix operand order checked by reading, not run on hardware
//
#include <hip/hip_runtime.h>
#include <hip/hip_bf16.h>

typedef __attribute__((ext_vector_type(16))) _Float16 v16h;
typedef __attribute__((ext_vector_type(8)))  _Float16 v8h;
typedef __attribute__((ext_vector_type(8)))  float    v8f;
typedef __attribute__((ext_vector_type(4)))  float    v4f;

// ---------------------------------------------------------------------------
// Fragment loaders for V_WMMA_F32_16X16X32_F16 (wave32).
// A (16x32, MxK): lane l holds row M=l&15; kbase=(l>>4)*8+32*kstep;
//   elements 0..7 -> K=kbase..kbase+7, elements 8..15 -> K=kbase+16..kbase+23.
// B (32x16, KxN): lane l holds column N=l&15 with identical K indexing, i.e.
//   loading row n of W^T (f16, [N][K]) gives the B fragment directly.
// C/D (16x16 f32, 8 VGPRs): lane l, reg r -> row r+8*(l>>4), col l&15.
// ---------------------------------------------------------------------------

__device__ __forceinline__ v16h make_a_f32(const float* __restrict__ row, int kbase) {
    v4f x0 = *(const v4f*)(row + kbase);
    v4f x1 = *(const v4f*)(row + kbase + 4);
    v4f y0 = *(const v4f*)(row + kbase + 16);
    v4f y1 = *(const v4f*)(row + kbase + 20);
    v16h a;
#pragma unroll
    for (int i = 0; i < 4; ++i) {
        a[i]      = (_Float16)x0[i];
        a[4 + i]  = (_Float16)x1[i];
        a[8 + i]  = (_Float16)y0[i];
        a[12 + i] = (_Float16)y1[i];
    }
    return a;
}

__device__ __forceinline__ v16h make_a_sum(const float* __restrict__ r1,
                                           const float* __restrict__ r2, int kbase) {
    v4f x0 = *(const v4f*)(r1 + kbase)      + *(const v4f*)(r2 + kbase);
    v4f x1 = *(const v4f*)(r1 + kbase + 4)  + *(const v4f*)(r2 + kbase + 4);
    v4f y0 = *(const v4f*)(r1 + kbase + 16) + *(const v4f*)(r2 + kbase + 16);
    v4f y1 = *(const v4f*)(r1 + kbase + 20) + *(const v4f*)(r2 + kbase + 20);
    v16h a;
#pragma unroll
    for (int i = 0; i < 4; ++i) {
        a[i]      = (_Float16)x0[i];
        a[4 + i]  = (_Float16)x1[i];
        a[8 + i]  = (_Float16)y0[i];
        a[12 + i] = (_Float16)y1[i];
    }
    return a;
}

__device__ __forceinline__ v16h make_a_pair(const float* __restrict__ p0,
                                            const float* __restrict__ p1) {
    v4f x0 = *(const v4f*)(p0);
    v4f x1 = *(const v4f*)(p0 + 4);
    v4f y0 = *(const v4f*)(p1);
    v4f y1 = *(const v4f*)(p1 + 4);
    v16h a;
#pragma unroll
    for (int i = 0; i < 4; ++i) {
        a[i]      = (_Float16)x0[i];
        a[4 + i]  = (_Float16)x1[i];
        a[8 + i]  = (_Float16)y0[i];
        a[12 + i] = (_Float16)y1[i];
    }
    return a;
}

__device__ __forceinline__ v16h make_b_f16(const _Float16* __restrict__ row, int kbase) {
    v8h x = *(const v8h*)(row + kbase);
    v8h y = *(const v8h*)(row + kbase + 16);
    v16h b;
#pragma unroll
    for (int i = 0; i < 8; ++i) { b[i] = x[i]; b[8 + i] = y[i]; }
    return b;
}

#define WMMA_F32_F16(a, b, c) \
    __builtin_amdgcn_wmma_f32_16x16x32_f16(false, (a), false, (b), (short)0, (c), false, false)

// ---------------------------------------------------------------------------
// CDNA5 async global->LDS DMA (tracked by ASYNCcnt). Per-lane 16B transfer.
// Generic shared pointer truncates to the wave-relative LDS byte offset
// (aperture mapping: low 32 bits of the address).
// ---------------------------------------------------------------------------
__device__ __forceinline__ void async_load_b128(const void* gsrc, void* ldst) {
    unsigned lds_off = (unsigned)(size_t)ldst;
    unsigned long long ga = (unsigned long long)(size_t)gsrc;
    asm volatile("global_load_async_to_lds_b128 %0, %1, off"
                 :: "v"(lds_off), "v"(ga) : "memory");
}
__device__ __forceinline__ void wait_asynccnt0() {
    asm volatile("s_wait_asynccnt 0x0" ::: "memory");
}

// Copy `halves` f16 elements (multiple of 8) from global to LDS, block-wide.
__device__ __forceinline__ void stage_weights_async(const _Float16* __restrict__ g,
                                                    _Float16* l, int halves) {
    int chunks = halves >> 3;  // 16B chunks
    for (int i = threadIdx.x; i < chunks; i += blockDim.x)
        async_load_b128(g + i * 8, l + i * 8);
}

// ---------------------------------------------------------------------------
// Utility kernels
// ---------------------------------------------------------------------------

__global__ void k_zero_f32(float* __restrict__ p, int n) {
    int i = blockIdx.x * blockDim.x + threadIdx.x;
    for (; i < n; i += gridDim.x * blockDim.x) p[i] = 0.f;
}

// dst[c*rows + r] = (f16) src[r*cols + c]   (W -> W^T in f16)
__global__ void k_transpose_f16(_Float16* __restrict__ dst, const float* __restrict__ src,
                                int rows, int cols) {
    int i = blockIdx.x * blockDim.x + threadIdx.x;
    int total = rows * cols;
    for (; i < total; i += gridDim.x * blockDim.x) {
        int r = i / cols, c = i - r * cols;
        dst[c * rows + r] = (_Float16)src[i];
    }
}

// ---------------------------------------------------------------------------
// WMMA GEMM: Cout[M,NC] = act( (A (+A2)) [M,K] @ B [K,NC] + bias )
// BT is f16 [NC][K]. grid.x = M/16, block = 256 (8 waves); wave handles
// column tiles wave, wave+8, ... All shape/fusion flags are compile-time so
// the inner loop is a branch-free, fully-unrolled load+wmma stream.
// ---------------------------------------------------------------------------
template <int K, int NC, bool HAS_A2, bool RELU>
__global__ void __launch_bounds__(256)
k_gemm(const float* __restrict__ A, const float* __restrict__ A2,
       const _Float16* __restrict__ BT, const float* __restrict__ bias,
       float* __restrict__ Cout) {
    int lane = threadIdx.x & 31;
    int wave = threadIdx.x >> 5;
    int laneLo = lane & 15, grp = lane >> 4;
    int mbase = blockIdx.x * 16;

    const float* arow  = A + (size_t)(mbase + laneLo) * K;
    const float* arow2 = HAS_A2 ? (A2 + (size_t)(mbase + laneLo) * K) : nullptr;
    __builtin_prefetch(arow, 0, 1);

    constexpr int NTILES = NC >> 4;
#pragma unroll
    for (int nt = wave; nt < NTILES; nt += 8) {
        const _Float16* brow = BT + (size_t)(nt * 16 + laneLo) * K;
        v8f c = {};
#pragma unroll
        for (int ks = 0; ks < K; ks += 32) {
            int kbase = ks + grp * 8;
            v16h a = HAS_A2 ? make_a_sum(arow, arow2, kbase) : make_a_f32(arow, kbase);
            v16h b = make_b_f16(brow, kbase);
            c = WMMA_F32_F16(a, b, c);
        }
        int col = nt * 16 + laneLo;
        float bc = bias[col];
#pragma unroll
        for (int r = 0; r < 8; ++r) {
            int row = mbase + r + 8 * grp;
            float v = c[r] + bc;
            if (RELU) v = fmaxf(v, 0.f);
            Cout[(size_t)row * NC + col] = v;
        }
    }
}

// ---------------------------------------------------------------------------
// Fused GINE edge step: e = ea@We+be ; msg = relu(h[src]+e) ; agg[dst] += msg
// One wave per 16-edge tile (8 waves / block). K=64, NC=128 (8 column tiles).
// WeT (128x64 f16 = 16KB) is async-DMA'd into LDS once per block.
// ---------------------------------------------------------------------------
__global__ void __launch_bounds__(256)
k_edge_gine(const float* __restrict__ ea, const _Float16* __restrict__ WeT,
            const float* __restrict__ be, const float* __restrict__ h,
            const int* __restrict__ ei, float* __restrict__ agg, int E_) {
    __shared__ __align__(16) _Float16 wlds[128 * 64];  // 16KB staged B

    stage_weights_async(WeT, wlds, 128 * 64);

    int lane = threadIdx.x & 31, wave = threadIdx.x >> 5;
    int laneLo = lane & 15, grp = lane >> 4;
    int ebase = (blockIdx.x * 8 + wave) * 16;

    const float* arow = ea + (size_t)(ebase + laneLo) * 64;
    __builtin_prefetch(arow, 0, 1);

    int srcs[8], dsts[8];
#pragma unroll
    for (int r = 0; r < 8; ++r) {
        int e = ebase + r + 8 * grp;
        srcs[r] = ei[e];
        dsts[r] = ei[E_ + e];
    }

    wait_asynccnt0();
    __syncthreads();

    v8f zero = {};
    v8f c[8];
#pragma unroll
    for (int nt = 0; nt < 8; ++nt) c[nt] = zero;

#pragma unroll
    for (int ks = 0; ks < 64; ks += 32) {
        int kbase = ks + grp * 8;
        v16h a = make_a_f32(arow, kbase);
#pragma unroll
        for (int nt = 0; nt < 8; ++nt) {
            v16h b = make_b_f16(wlds + (nt * 16 + laneLo) * 64, kbase);
            c[nt] = WMMA_F32_F16(a, b, c[nt]);
        }
    }

#pragma unroll
    for (int nt = 0; nt < 8; ++nt) {
        int col = nt * 16 + laneLo;
        float bc = be[col];
#pragma unroll
        for (int r = 0; r < 8; ++r) {
            float v = fmaxf(c[nt][r] + bc + h[(size_t)srcs[r] * 128 + col], 0.f);
            atomicAdd(&agg[(size_t)dsts[r] * 128 + col], v);
        }
    }
}

// ---------------------------------------------------------------------------
// BatchNorm stats: column sums / sums of squares over z [M,128]
// ---------------------------------------------------------------------------
__global__ void k_colstats(const float* __restrict__ z, float* __restrict__ sums,
                           float* __restrict__ sumsq, int M) {
    int col = threadIdx.x;  // blockDim.x == 128
    float s = 0.f, sq = 0.f;
    for (int r = blockIdx.x; r < M; r += gridDim.x) {
        float v = z[(size_t)r * 128 + col];
        s += v;
        sq += v * v;
    }
    atomicAdd(&sums[col], s);
    atomicAdd(&sumsq[col], sq);
}

__global__ void k_bn_relu(const float* __restrict__ z, const float* __restrict__ sums,
                          const float* __restrict__ sumsq, const float* __restrict__ gamma,
                          const float* __restrict__ beta, float* __restrict__ hout, int M) {
    int total = M * 128;
    float invM = 1.f / (float)M;
    for (int i = blockIdx.x * blockDim.x + threadIdx.x; i < total;
         i += gridDim.x * blockDim.x) {
        int col = i & 127;
        float mu  = sums[col] * invM;
        float var = sumsq[col] * invM - mu * mu;
        float v = (z[i] - mu) * rsqrtf(var + 1e-5f) * gamma[col] + beta[col];
        hout[i] = fmaxf(v, 0.f);
    }
}

// ---------------------------------------------------------------------------
// Fully fused edge head per 16-edge tile:
//   y1 = relu(concat(h[src],h[dst],ea) @ Wh1 + bh1)   K=320, NC=128 (WMMA)
//   y2 = relu(y1 @ Wh2 + bh2)                         K=128, NC=64  (WMMA, via LDS)
//   out = y2 @ Wh3 + bh3                              K=64,  NC=4   (VALU)
// Wh2T (16KB) is async-DMA'd into LDS, overlapping GEMM1.
// ---------------------------------------------------------------------------
__global__ void __launch_bounds__(256)
k_edge_head(const float* __restrict__ h, const float* __restrict__ ea,
            const int* __restrict__ ei, const _Float16* __restrict__ Wh1T,
            const float* __restrict__ bh1, const _Float16* __restrict__ Wh2T,
            const float* __restrict__ bh2, const float* __restrict__ Wh3,
            const float* __restrict__ bh3, float* __restrict__ out, int E_) {
    __shared__ __align__(16) _Float16 yl[8][2048];   // per-wave 16x128 f16 tile (32KB)
    __shared__ __align__(16) _Float16 w2l[64 * 128]; // 16KB staged Wh2T

    stage_weights_async(Wh2T, w2l, 64 * 128);        // overlaps with GEMM1

    int lane = threadIdx.x & 31, wave = threadIdx.x >> 5;
    int laneLo = lane & 15, grp = lane >> 4;
    int ebase = (blockIdx.x * 8 + wave) * 16;
    int e0 = ebase + laneLo;

    const float* hs = h + (size_t)ei[e0] * 128;
    const float* hd = h + (size_t)ei[E_ + e0] * 128;
    const float* er = ea + (size_t)e0 * 64;
    __builtin_prefetch(er, 0, 1);
    __builtin_prefetch(hs, 0, 1);

    // ---- GEMM1: [16 x 320] @ [320 x 128] ----
    v8f zero = {};
    v8f c[8];
#pragma unroll
    for (int nt = 0; nt < 8; ++nt) c[nt] = zero;

#pragma unroll
    for (int ks = 0; ks < 320; ks += 32) {
        int k0 = ks + grp * 8;
        int k1 = k0 + 16;
        const float* p0 = (k0 < 128) ? (hs + k0) : ((k0 < 256) ? (hd + k0 - 128) : (er + k0 - 256));
        const float* p1 = (k1 < 128) ? (hs + k1) : ((k1 < 256) ? (hd + k1 - 128) : (er + k1 - 256));
        v16h a = make_a_pair(p0, p1);
#pragma unroll
        for (int nt = 0; nt < 8; ++nt) {
            v16h b = make_b_f16(Wh1T + (size_t)(nt * 16 + laneLo) * 320, k0);
            c[nt] = WMMA_F32_F16(a, b, c[nt]);
        }
    }

    // relu + bias, spill tile to LDS as f16 (C-layout -> row-major)
    _Float16* my = &yl[wave][0];
#pragma unroll
    for (int nt = 0; nt < 8; ++nt) {
        int col = nt * 16 + laneLo;
        float bc = bh1[col];
#pragma unroll
        for (int r = 0; r < 8; ++r) {
            my[(r + 8 * grp) * 128 + col] = (_Float16)fmaxf(c[nt][r] + bc, 0.f);
        }
    }
    wait_asynccnt0();   // Wh2T DMA landed in LDS
    __syncthreads();

    // ---- GEMM2: [16 x 128] @ [128 x 64], A from LDS, B from staged LDS ----
    v8f c2[4];
#pragma unroll
    for (int nt = 0; nt < 4; ++nt) c2[nt] = zero;

    const _Float16* arow2 = my + laneLo * 128;
#pragma unroll
    for (int ks = 0; ks < 128; ks += 32) {
        int kbase = ks + grp * 8;
        v16h a = make_b_f16(arow2, kbase);  // same layout loader (f16 rows)
#pragma unroll
        for (int nt = 0; nt < 4; ++nt) {
            v16h b = make_b_f16(w2l + (nt * 16 + laneLo) * 128, kbase);
            c2[nt] = WMMA_F32_F16(a, b, c2[nt]);
        }
    }
    __syncthreads();

    // relu + bias, store y2 (16x64 f32) into same LDS region
    float* y2l = (float*)my;
#pragma unroll
    for (int nt = 0; nt < 4; ++nt) {
        int col = nt * 16 + laneLo;
        float bc = bh2[col];
#pragma unroll
        for (int r = 0; r < 8; ++r) {
            y2l[(r + 8 * grp) * 64 + col] = fmaxf(c2[nt][r] + bc, 0.f);
        }
    }
    __syncthreads();

    // ---- GEMM3: [16 x 64] @ [64 x 4] in VALU (64 outputs, 2 per lane) ----
    for (int oi = lane; oi < 64; oi += 32) {
        int row = oi >> 2, cc = oi & 3;
        float acc = bh3[cc];
        const float* yr = y2l + row * 64;
#pragma unroll
        for (int k = 0; k < 64; ++k) acc += yr[k] * Wh3[k * 4 + cc];
        out[(size_t)(ebase + row) * 4 + cc] = acc;
    }
}

// ---------------------------------------------------------------------------
// Host orchestration
// ---------------------------------------------------------------------------
extern "C" void kernel_launch(void* const* d_in, const int* in_sizes, int n_in,
                              void* d_out, int out_size, void* d_ws, size_t ws_size,
                              hipStream_t stream) {
    const float* x    = (const float*)d_in[0];
    const int*   ei   = (const int*)d_in[1];
    const float* ea   = (const float*)d_in[2];
    const float* Wp   = (const float*)d_in[3];
    const float* bp   = (const float*)d_in[4];
    const float* We   = (const float*)d_in[5];
    const float* be   = (const float*)d_in[6];
    const float* W1   = (const float*)d_in[7];
    const float* b1   = (const float*)d_in[8];
    const float* W2   = (const float*)d_in[9];
    const float* b2   = (const float*)d_in[10];
    const float* gam  = (const float*)d_in[11];
    const float* bet  = (const float*)d_in[12];
    const float* Wh1  = (const float*)d_in[13];
    const float* bh1  = (const float*)d_in[14];
    const float* Wh2  = (const float*)d_in[15];
    const float* bh2  = (const float*)d_in[16];
    const float* Wh3  = (const float*)d_in[17];
    const float* bh3  = (const float*)d_in[18];
    float* out = (float*)d_out;

    const int N = in_sizes[0] / 256;   // 10000 (multiple of 16)
    const int E = in_sizes[2] / 64;    // 640000 (multiple of 128)

    // Workspace layout (floats, then f16 transposed weights)
    float* wH   = (float*)d_ws;                 // [N,128] node features
    float* wAgg = wH   + (size_t)N * 128;       // [N,128] aggregation
    float* wT   = wAgg + (size_t)N * 128;       // [N,256] MLP hidden
    float* wZ   = wT   + (size_t)N * 256;       // [N,128] pre-BN
    float* wSum = wZ   + (size_t)N * 128;       // [128] + [128] stats
    float* wSq  = wSum + 128;
    _Float16* WpT  = (_Float16*)(wSq + 128);    // [128][256]
    _Float16* WeT  = WpT  + 128 * 256;          // 3 x [128][64]
    _Float16* W1T  = WeT  + 3 * 128 * 64;       // 3 x [256][128]
    _Float16* W2T  = W1T  + 3 * 256 * 128;      // 3 x [128][256]
    _Float16* Wh1T = W2T  + 3 * 128 * 256;      // [128][320]
    _Float16* Wh2T = Wh1T + 128 * 320;          // [64][128]

    // Weight transpose + f16 convert
    k_transpose_f16<<<64, 256, 0, stream>>>(WpT, Wp, 256, 128);
    for (int l = 0; l < 3; ++l) {
        k_transpose_f16<<<16, 256, 0, stream>>>(WeT + l * 128 * 64, We + l * 64 * 128, 64, 128);
        k_transpose_f16<<<64, 256, 0, stream>>>(W1T + l * 256 * 128, W1 + l * 128 * 256, 128, 256);
        k_transpose_f16<<<64, 256, 0, stream>>>(W2T + l * 128 * 256, W2 + l * 256 * 128, 256, 128);
    }
    k_transpose_f16<<<64, 256, 0, stream>>>(Wh1T, Wh1, 320, 128);
    k_transpose_f16<<<16, 256, 0, stream>>>(Wh2T, Wh2, 128, 64);

    // h = relu(x @ Wp + bp)
    k_gemm<256, 128, false, true><<<N / 16, 256, 0, stream>>>(x, nullptr, WpT, bp, wH);

    for (int l = 0; l < 3; ++l) {
        k_zero_f32<<<512, 256, 0, stream>>>(wAgg, N * 128);
        k_zero_f32<<<1, 256, 0, stream>>>(wSum, 256);  // sums + sumsq contiguous
        // agg = segment_sum(relu(h[src] + ea@We+be), dst)
        k_edge_gine<<<E / 128, 256, 0, stream>>>(ea, WeT + l * 128 * 64, be + l * 128,
                                                 wH, ei, wAgg, E);
        // t = relu((agg + h) @ W1 + b1)
        k_gemm<128, 256, true, true><<<N / 16, 256, 0, stream>>>(wAgg, wH, W1T + l * 256 * 128,
                                                                 b1 + l * 256, wT);
        // z = t @ W2 + b2
        k_gemm<256, 128, false, false><<<N / 16, 256, 0, stream>>>(wT, nullptr,
                                                                   W2T + l * 128 * 256,
                                                                   b2 + l * 128, wZ);
        // batchnorm stats + apply + relu -> h
        k_colstats<<<80, 128, 0, stream>>>(wZ, wSum, wSq, N);
        k_bn_relu<<<1024, 256, 0, stream>>>(wZ, wSum, wSq, gam + l * 128, bet + l * 128, wH, N);
    }

    // Fused edge head: concat -> 3 GEMMs -> out [E,4]
    k_edge_head<<<E / 128, 256, 0, stream>>>(wH, ea, ei, Wh1T, bh1, Wh2T, bh2,
                                             Wh3, bh3, out, E);
}
